// MinkUNet_13288628814177
// MI455X (gfx1250) — matrix operand order repaired and checked by reading, not compiled, not run on hardware
//
#include <hip/hip_runtime.h>

// ---------------------------------------------------------------------------
// MinkUNet (dense 48^3) for MI455X / gfx1250.
// All convs = implicit GEMM on V_WMMA_F32_16X16X32_F16 (wave32).
// Activations/weights f16 (channels-last, Cin padded to mult of 32),
// f32 accumulate, training-mode BN re-normalizes every layer.
// Block = 8 waves x (16 voxels each) sharing one 32-channel N-tile whose
// weights are staged per-tap into LDS, double-buffered and (when the
// toolchain exposes it) via GLOBAL_LOAD_ASYNC_TO_LDS_B128 + s_wait_asynccnt.
// ---------------------------------------------------------------------------

typedef _Float16 half8  __attribute__((ext_vector_type(8)));
typedef _Float16 half16 __attribute__((ext_vector_type(16)));
typedef float    float8 __attribute__((ext_vector_type(8)));
typedef int      int4v  __attribute__((ext_vector_type(4)));

#define WPB 8  // waves per 256-thread block

#if defined(__has_builtin)
# if __has_builtin(__builtin_amdgcn_global_load_async_to_lds_b128) && \
     __has_builtin(__builtin_amdgcn_s_wait_asynccnt)
#  define USE_ASYNC_LDS 1
# endif
#endif
#ifndef USE_ASYNC_LDS
# define USE_ASYNC_LDS 0
#endif

#if USE_ASYNC_LDS
typedef __attribute__((address_space(1))) int4v as1_int4;  // global (b128)
typedef __attribute__((address_space(3))) int4v as3_int4;  // LDS (b128)
// 16B global -> LDS async copy (ASYNCcnt-tracked). AS1 = flat address;
// AS3 = low 32 bits of the generic LDS address.
__device__ inline void async_copy16(const void* g, void* l) {
  __builtin_amdgcn_global_load_async_to_lds_b128(
      (as1_int4*)(unsigned long long)g,
      (as3_int4*)(unsigned int)(unsigned long long)l, 0, 0);
}
#endif

__device__ inline half16 cat8(half8 lo, half8 hi) {
  return __builtin_shufflevector(lo, hi, 0,1,2,3,4,5,6,7,8,9,10,11,12,13,14,15);
}

// Generic conv / deconv implicit GEMM, specialized at compile time.
//   MODE 0: conv, output voxel (d,h,w) reads input (d*stride+t-pad, ...)
//   MODE 1: deconv ks2/s2 tap-scatter: M-tiles over the COARSE grid, tap from
//           blockIdx; fine voxel (2d+td,2h+th,2w+tw) written exactly once.
// act [vox][CinP] f16, wt [tap][co][CinP] f16, out [vox][Cout] f32.
// One wave = 16 voxels x 32 channels (2 accumulators); 8 waves per block
// share the double-buffered B tile staged in LDS.
template<int KS, int MODE>
__global__ __launch_bounds__(256) void conv_wmma_t(
    const _Float16* __restrict__ act, const _Float16* __restrict__ wt,
    float* __restrict__ out, int Gin, int Gout, int CinP, int Cout,
    int stride, int pad, int nMtB)
{
  extern __shared__ _Float16 Bs[];   // 2 x [32][CinP+8] padded (bank stagger)
  const int tid  = threadIdx.x;
  const int lane = tid & 31;
  const int wid  = tid >> 5;

  const int nNt = Cout >> 5;
  int bx = blockIdx.x;
  const int mtb = bx % nMtB; bx /= nMtB;
  const int nt  = bx % nNt;  bx /= nNt;
  const int tapSel = bx;                    // 0 when MODE==0

  const int  Gm    = (MODE == 1) ? Gin : Gout;   // grid the M-tiles cover
  const long NvoxM = (long)Gm * Gm * Gm;
  const int  mt    = mtb * WPB + wid;
  const int  m     = lane & 15;             // A-matrix row within tile
  const int  khalf = lane >> 4;             // K-half (ISA A layout)

  long vox = (long)mt * 16 + m;
  const bool mvalid = vox < NvoxM;
  long vv = mvalid ? vox : 0;
  int wm = (int)(vv % Gm); int tq = (int)(vv / Gm);
  int hm = tq % Gm;        int dm = tq / Gm;

  const int CinPad    = CinP + 8;
  const int bufStride = 32 * CinPad;
  const int nK        = CinP >> 5;
  const int ntap      = (MODE == 1) ? 1 : KS * KS * KS;

  // Cooperative stage of one 32xCinP weight tile into LDS buffer `buf`.
  auto stage = [&](int buf, int tt) {
    const _Float16* wsrc = wt + ((long)tt * Cout + (long)nt * 32) * (long)CinP;
    _Float16* dstB = Bs + buf * bufStride;
    const int chunks = CinP >> 3;                // half8 (16B) chunks per row
    for (int i = tid; i < 32 * chunks; i += 256) {
      int co = i / chunks, k8 = i % chunks;
#if USE_ASYNC_LDS
      async_copy16(wsrc + (long)co * CinP + k8 * 8, dstB + co * CinPad + k8 * 8);
#else
      *(half8*)(dstB + co * CinPad + k8 * 8) =
          *(const half8*)(wsrc + (long)co * CinP + k8 * 8);
#endif
    }
  };

  float8 acc0 = {}; float8 acc1 = {};

  stage(0, (MODE == 1) ? tapSel : 0);            // prologue: tap 0 -> buf 0
#if USE_ASYNC_LDS
  __builtin_amdgcn_s_wait_asynccnt(0);
#endif
  __syncthreads();

  for (int t = 0; t < ntap; ++t) {
    const int tt = (MODE == 1) ? tapSel : t;
    const int td = tt / (KS * KS), th = (tt / KS) % KS, tw = tt % KS;

    if (t + 1 < ntap) stage((t + 1) & 1, t + 1); // prefetch next tap (overlaps)

    int din, hin, win; bool avalid = mvalid;
    if (MODE == 1) { din = dm; hin = hm; win = wm; }
    else {
      din = dm * stride + td - pad;
      hin = hm * stride + th - pad;
      win = wm * stride + tw - pad;
      avalid = avalid && (unsigned)din < (unsigned)Gin &&
               (unsigned)hin < (unsigned)Gin && (unsigned)win < (unsigned)Gin;
    }
    const long abase = (((long)din * Gin + hin) * Gin + win) * (long)CinP;
    __builtin_prefetch(act + (avalid ? abase : 0), 0, 1);  // global_prefetch_b8

    const _Float16* Bbuf = Bs + (t & 1) * bufStride;
    for (int kc = 0; kc < nK; ++kc) {
      const int kb = kc << 5;
      half8 z = {}; half8 a0 = z, a1 = z;
      if (avalid) {                               // zero-pad OOB (SAME conv)
        const _Float16* ap = act + abase + kb + khalf * 8;
        a0 = *(const half8*)ap;
        a1 = *(const half8*)(ap + 16);
      }
      half16 afrag = cat8(a0, a1);

      // B fragments from LDS: lane = out-channel column, K = 8h..8h+7, 16+8h..16+8h+7
      const _Float16* bp0 = Bbuf + (lane & 15) * CinPad + kb + khalf * 8;
      half16 bf0 = cat8(*(const half8*)bp0, *(const half8*)(bp0 + 16));
      const _Float16* bp1 = bp0 + 16 * CinPad;
      half16 bf1 = cat8(*(const half8*)bp1, *(const half8*)(bp1 + 16));

      acc0 = __builtin_amdgcn_wmma_f32_16x16x32_f16(false, afrag, false, bf0,
                                                    (short)0, acc0, false, false);
      acc1 = __builtin_amdgcn_wmma_f32_16x16x32_f16(false, afrag, false, bf1,
                                                    (short)0, acc1, false, false);
    }
#if USE_ASYNC_LDS
    __builtin_amdgcn_s_wait_asynccnt(0);         // next tap staged
#endif
    __syncthreads();
  }

  // D layout: VGPR r, lane l -> M = r + 8*(l/16), N = l%16
  const int col = nt * 32 + (lane & 15);
  for (int r = 0; r < 8; ++r) {
    long mv = (long)mt * 16 + r + 8 * khalf;
    if (mv < NvoxM) {
      int w2 = (int)(mv % Gm); int q = (int)(mv / Gm);
      int h2 = q % Gm;         int d2 = q / Gm;
      long ovox;
      if (MODE == 1) {
        int tdd = tapSel >> 2, thh = (tapSel >> 1) & 1, tww = tapSel & 1;
        ovox = (((long)(2 * d2 + tdd) * Gout) + (2 * h2 + thh)) * Gout + (2 * w2 + tww);
      } else {
        ovox = mv;
      }
      out[ovox * Cout + col]      = acc0[r];
      out[ovox * Cout + col + 16] = acc1[r];
    }
  }
}

// Per-channel sum / sumsq for training-mode BN. One workgroup per channel.
__global__ __launch_bounds__(256) void bn_stats_k(
    const float* __restrict__ x, float* __restrict__ stats, int Nvox, int C)
{
  __shared__ float ssum[256], ssq[256];
  const int c = blockIdx.x;
  float s = 0.f, q = 0.f;
  for (int v = threadIdx.x; v < Nvox; v += 256) {
    float t = x[(long)v * C + c]; s += t; q += t * t;
  }
  ssum[threadIdx.x] = s; ssq[threadIdx.x] = q;
  __syncthreads();
  for (int o = 128; o > 0; o >>= 1) {
    if (threadIdx.x < o) {
      ssum[threadIdx.x] += ssum[threadIdx.x + o];
      ssq[threadIdx.x]  += ssq[threadIdx.x + o];
    }
    __syncthreads();
  }
  if (threadIdx.x == 0) { stats[2 * c] = ssum[0]; stats[2 * c + 1] = ssq[0]; }
}

// BN normalize + affine, optional residual add (f16 src), optional ReLU,
// write f16 at [v*dstStride + dstOff + c]. mode: 0=bn, 1=bn+relu, 2=bn+sc+relu
__global__ __launch_bounds__(256) void bn_apply_k(
    const float* __restrict__ src, _Float16* __restrict__ dst,
    const float* __restrict__ g, const float* __restrict__ b,
    const float* __restrict__ stats, const _Float16* __restrict__ sc,
    int scStride, int Nvox, int C, int dstStride, int dstOff, int mode, long total)
{
  long i = (long)blockIdx.x * 256 + threadIdx.x;
  if (i >= total) return;
  int c = (int)(i % C); long v = i / C;
  float invN = 1.0f / (float)Nvox;
  float mean = stats[2 * c] * invN;
  float var  = stats[2 * c + 1] * invN - mean * mean;
  float y = (src[i] - mean) * rsqrtf(var + 1e-5f) * g[c] + b[c];
  if (mode == 2) y += (float)sc[v * (long)scStride + c];
  if (mode != 0) y = fmaxf(y, 0.0f);
  dst[v * (long)dstStride + dstOff + c] = (_Float16)y;
}

// Copy skip features into the concat buffer (channel offset).
__global__ __launch_bounds__(256) void copy_ch_k(
    const _Float16* __restrict__ src, int srcStride, _Float16* __restrict__ dst,
    int dstStride, int dstOff, long Nvox, int C)
{
  long i = (long)blockIdx.x * 256 + threadIdx.x;
  if (i >= Nvox * C) return;
  int c = (int)(i % C); long v = i / C;
  dst[v * (long)dstStride + dstOff + c] = src[v * (long)srcStride + c];
}

// NCDHW f32 input (3 ch) -> [vox][32] f16, channels 3..31 zero.
__global__ __launch_bounds__(256) void cvt_in_k(
    const float* __restrict__ x, _Float16* __restrict__ dst, int Nvox)
{
  long i = (long)blockIdx.x * 256 + threadIdx.x;
  if (i >= (long)Nvox * 32) return;
  int c = (int)(i & 31); long v = i >> 5;
  float val = (c < 3) ? x[(long)c * Nvox + v] : 0.0f;
  dst[i] = (_Float16)val;
}

// OIDHW f32 weights -> [tap][co][CinP] f16 (ci >= Cin zero-padded).
__global__ __launch_bounds__(256) void cvt_w_k(
    const float* __restrict__ w, _Float16* __restrict__ dst,
    int Co, int Ci, int CinP, int kt, long total)
{
  long i = (long)blockIdx.x * 256 + threadIdx.x;
  if (i >= total) return;
  int ci = (int)(i % CinP); long r = i / CinP;
  int co = (int)(r % Co);   int t = (int)(r / Co);
  float v = (ci < Ci) ? w[((long)co * Ci + ci) * kt + t] : 0.0f;
  dst[i] = (_Float16)v;
}

// Per-voxel head MLP: 96 -> 20 (LeakyReLU 0.1) -> 3 (tanh), f32 out.
__global__ __launch_bounds__(256) void head_k(
    const _Float16* __restrict__ act, const float* __restrict__ w1,
    const float* __restrict__ b1, const float* __restrict__ w2,
    const float* __restrict__ b2, float* __restrict__ out, int Nvox, int C)
{
  int v = blockIdx.x * 256 + threadIdx.x;
  if (v >= Nvox) return;
  float h[20];
  for (int j = 0; j < 20; ++j) {
    float s = b1[j];
    for (int c = 0; c < C; ++c) s += (float)act[(long)v * C + c] * w1[c * 20 + j];
    h[j] = (s >= 0.f) ? s : 0.1f * s;
  }
  for (int j = 0; j < 3; ++j) {
    float s = b2[j];
    for (int k = 0; k < 20; ++k) s += h[k] * w2[k * 3 + j];
    out[(long)v * 3 + j] = tanhf(s);
  }
}

// ---------------------------------------------------------------------------
extern "C" void kernel_launch(void* const* d_in, const int* in_sizes, int n_in,
                              void* d_out, int out_size, void* d_ws, size_t ws_size,
                              hipStream_t stream) {
  (void)in_sizes; (void)n_in; (void)out_size; (void)ws_size;

  // ---- unpack params in JAX pytree (sorted dict key) flatten order ----
  int ii = 0;
  auto nxt = [&]() { return (const float*)d_in[ii++]; };
  struct CP { const float* b; const float* g; const float* w; };
  auto cp = [&]() { CP c; c.b = nxt(); c.g = nxt(); c.w = nxt(); return c; };

  const float* head_b1 = nxt(); const float* head_b2 = nxt();
  const float* head_w1 = nxt(); const float* head_w2 = nxt();
  CP s1dn=cp(), s1r1c1=cp(), s1r1c2=cp(), s1r2c1=cp(), s1r2c2=cp();
  CP s2dn=cp(), s2r1c1=cp(), s2r1c2=cp(), s2r1ds=cp(), s2r2c1=cp(), s2r2c2=cp();
  CP s3dn=cp(), s3r1c1=cp(), s3r1c2=cp(), s3r1ds=cp(), s3r2c1=cp(), s3r2c2=cp();
  CP s4dn=cp(), s4r1c1=cp(), s4r1c2=cp(), s4r1ds=cp(), s4r2c1=cp(), s4r2c2=cp();
  CP stem0=cp(), stem1=cp();
  CP u1dc=cp(), u1r1c1=cp(), u1r1c2=cp(), u1r1ds=cp(), u1r2c1=cp(), u1r2c2=cp();
  CP u2dc=cp(), u2r1c1=cp(), u2r1c2=cp(), u2r1ds=cp(), u2r2c1=cp(), u2r2c2=cp();
  CP u3dc=cp(), u3r1c1=cp(), u3r1c2=cp(), u3r1ds=cp(), u3r2c1=cp(), u3r2c2=cp();
  CP u4dc=cp(), u4r1c1=cp(), u4r1c2=cp(), u4r1ds=cp(), u4r2c1=cp(), u4r2c2=cp();
  const float* x = nxt();

  // ---- workspace bump allocator ----
  char* ws = (char*)d_ws; size_t off = 0;
  auto alloc = [&](size_t bytes) -> void* {
    void* p = ws + off; off = (off + bytes + 255) & ~(size_t)255; return p;
  };

  const int V48 = 48*48*48, V24 = 24*24*24, V12 = 12*12*12, V6 = 6*6*6, V3 = 27;
  float*    F   = (float*)   alloc((size_t)V48 * 96  * sizeof(float)); // conv f32 out
  _Float16* SA  = (_Float16*)alloc((size_t)V48 * 128 * 2);  // stage-in / concat
  _Float16* SB  = (_Float16*)alloc((size_t)V48 * 96  * 2);  // shortcut (ds out)
  _Float16* SCb = (_Float16*)alloc((size_t)V48 * 96  * 2);  // res h1
  _Float16* SD  = (_Float16*)alloc((size_t)V48 * 96  * 2);  // r1 out
  _Float16* SE  = (_Float16*)alloc((size_t)V48 * 96  * 2);  // stage out
  _Float16* A0  = (_Float16*)alloc((size_t)V48 * 32  * 2);  // padded input
  _Float16* X0  = (_Float16*)alloc((size_t)V48 * 32  * 2);
  _Float16* X1  = (_Float16*)alloc((size_t)V24 * 32  * 2);
  _Float16* X2  = (_Float16*)alloc((size_t)V12 * 64  * 2);
  _Float16* X3  = (_Float16*)alloc((size_t)V6  * 128 * 2);
  _Float16* X4  = (_Float16*)alloc((size_t)V3  * 256 * 2);
  float* stats  = (float*)   alloc(512 * 2 * sizeof(float));

  // ---- weight conversion to [tap][co][CinP] f16 ----
  struct WT { const _Float16* p; int CinP, Co, ks; };
  auto mkw = [&](const CP& c, int Ci, int Co, int ks) {
    int kt = ks * ks * ks; int CinP = (Ci < 32) ? 32 : Ci;
    long n = (long)kt * Co * CinP;
    _Float16* dst = (_Float16*)alloc((size_t)n * 2);
    cvt_w_k<<<(int)((n + 255) / 256), 256, 0, stream>>>(c.w, dst, Co, Ci, CinP, kt, n);
    WT w; w.p = dst; w.CinP = CinP; w.Co = Co; w.ks = ks; return w;
  };

  WT wstem0 = mkw(stem0, 3, 32, 3),  wstem1 = mkw(stem1, 32, 32, 3);
  WT ws1dn = mkw(s1dn,32,32,2), ws1r1c1 = mkw(s1r1c1,32,32,3), ws1r1c2 = mkw(s1r1c2,32,32,3),
     ws1r2c1 = mkw(s1r2c1,32,32,3), ws1r2c2 = mkw(s1r2c2,32,32,3);
  WT ws2dn = mkw(s2dn,32,32,2), ws2r1c1 = mkw(s2r1c1,32,64,3), ws2r1c2 = mkw(s2r1c2,64,64,3),
     ws2r1ds = mkw(s2r1ds,32,64,1), ws2r2c1 = mkw(s2r2c1,64,64,3), ws2r2c2 = mkw(s2r2c2,64,64,3);
  WT ws3dn = mkw(s3dn,64,64,2), ws3r1c1 = mkw(s3r1c1,64,128,3), ws3r1c2 = mkw(s3r1c2,128,128,3),
     ws3r1ds = mkw(s3r1ds,64,128,1), ws3r2c1 = mkw(s3r2c1,128,128,3), ws3r2c2 = mkw(s3r2c2,128,128,3);
  WT ws4dn = mkw(s4dn,128,128,2), ws4r1c1 = mkw(s4r1c1,128,256,3), ws4r1c2 = mkw(s4r1c2,256,256,3),
     ws4r1ds = mkw(s4r1ds,128,256,1), ws4r2c1 = mkw(s4r2c1,256,256,3), ws4r2c2 = mkw(s4r2c2,256,256,3);
  WT wu1dc = mkw(u1dc,256,256,2), wu1r1c1 = mkw(u1r1c1,384,256,3), wu1r1c2 = mkw(u1r1c2,256,256,3),
     wu1r1ds = mkw(u1r1ds,384,256,1), wu1r2c1 = mkw(u1r2c1,256,256,3), wu1r2c2 = mkw(u1r2c2,256,256,3);
  WT wu2dc = mkw(u2dc,256,128,2), wu2r1c1 = mkw(u2r1c1,192,128,3), wu2r1c2 = mkw(u2r1c2,128,128,3),
     wu2r1ds = mkw(u2r1ds,192,128,1), wu2r2c1 = mkw(u2r2c1,128,128,3), wu2r2c2 = mkw(u2r2c2,128,128,3);
  WT wu3dc = mkw(u3dc,128,96,2), wu3r1c1 = mkw(u3r1c1,128,96,3), wu3r1c2 = mkw(u3r1c2,96,96,3),
     wu3r1ds = mkw(u3r1ds,128,96,1), wu3r2c1 = mkw(u3r2c1,96,96,3), wu3r2c2 = mkw(u3r2c2,96,96,3);
  WT wu4dc = mkw(u4dc,96,96,2), wu4r1c1 = mkw(u4r1c1,128,96,3), wu4r1c2 = mkw(u4r1c2,96,96,3),
     wu4r1ds = mkw(u4r1ds,128,96,1), wu4r2c1 = mkw(u4r2c1,96,96,3), wu4r2c2 = mkw(u4r2c2,96,96,3);

  // ---- layer launchers ----
  auto conv = [&](const _Float16* a, const WT& w, int Gin, int Gout,
                  int stride, int pad, int mode) {
    int Gm = (mode == 1) ? Gin : Gout;
    long nvox = (long)Gm * Gm * Gm;
    int nMt  = (int)((nvox + 15) / 16);
    int nMtB = (nMt + WPB - 1) / WPB;
    int nNt  = w.Co / 32;
    int blocks = nMtB * nNt * ((mode == 1) ? 8 : 1);
    size_t lds = (size_t)2 * 32 * (w.CinP + 8) * sizeof(_Float16);
    if (mode == 1)
      conv_wmma_t<2,1><<<blocks, 256, lds, stream>>>(a, w.p, F, Gin, Gout,
                                                     w.CinP, w.Co, stride, pad, nMtB);
    else if (w.ks == 3)
      conv_wmma_t<3,0><<<blocks, 256, lds, stream>>>(a, w.p, F, Gin, Gout,
                                                     w.CinP, w.Co, stride, pad, nMtB);
    else if (w.ks == 2)
      conv_wmma_t<2,0><<<blocks, 256, lds, stream>>>(a, w.p, F, Gin, Gout,
                                                     w.CinP, w.Co, stride, pad, nMtB);
    else
      conv_wmma_t<1,0><<<blocks, 256, lds, stream>>>(a, w.p, F, Gin, Gout,
                                                     w.CinP, w.Co, stride, pad, nMtB);
  };
  auto bn = [&](const CP& p, long Nv, int C, _Float16* dst, int dstStride,
                int dstOff, int mode, const _Float16* sc, int scStride) {
    bn_stats_k<<<C, 256, 0, stream>>>(F, stats, (int)Nv, C);
    long tot = Nv * C;
    bn_apply_k<<<(int)((tot + 255) / 256), 256, 0, stream>>>(
        F, dst, p.g, p.b, stats, sc, scStride, (int)Nv, C, dstStride, dstOff, mode, tot);
  };
  // ResidualBlock: relu(bn(conv3)), bn(conv3), (+1x1 conv-bn shortcut), relu
  auto res = [&](const _Float16* in, int G, int Ci, int Co,
                 const WT& wc1, const CP& c1, const WT& wc2, const CP& c2,
                 const WT* wds, const CP* dsc, _Float16* outb) {
    long Nv = (long)G * G * G;
    const _Float16* sc; int scStride;
    if (wds) {
      conv(in, *wds, G, G, 1, 0, 0);
      bn(*dsc, Nv, Co, SB, Co, 0, 0, nullptr, 0);
      sc = SB; scStride = Co;
    } else { sc = in; scStride = Ci; }
    conv(in, wc1, G, G, 1, 1, 0);
    bn(c1, Nv, Co, SCb, Co, 0, 1, nullptr, 0);
    conv(SCb, wc2, G, G, 1, 1, 0);
    bn(c2, Nv, Co, outb, Co, 0, 2, sc, scStride);
  };
  // Encoder stage: down(ks2,s2,VALID)+bn+relu, res, res -> skip buffer
  auto enc = [&](const _Float16* xin, int Gin, int Ci, int Co,
                 const WT& wdn, const CP& dn,
                 const WT& w11, const CP& c11, const WT& w12, const CP& c12,
                 const WT* w1ds, const CP* c1ds,
                 const WT& w21, const CP& c21, const WT& w22, const CP& c22,
                 _Float16* xout) {
    int Go = Gin / 2; long Nv = (long)Go * Go * Go;
    conv(xin, wdn, Gin, Go, 2, 0, 0);
    bn(dn, Nv, Ci, SA, Ci, 0, 1, nullptr, 0);
    res(SA, Go, Ci, Co, w11, c11, w12, c12, w1ds, c1ds, SD);
    res(SD, Go, Co, Co, w21, c21, w22, c22, nullptr, nullptr, xout);
  };
  // Decoder stage: deconv(ks2,s2)+bn+relu, concat skip, res(ds), res
  auto dec = [&](const _Float16* y, int Gc, const _Float16* skip, int Cskip,
                 const WT& wdc, const CP& dcp,
                 const WT& w11, const CP& c11, const WT& w12, const CP& c12,
                 const WT& w1ds, const CP& c1ds,
                 const WT& w21, const CP& c21, const WT& w22, const CP& c22,
                 _Float16* outb) {
    int Gf = Gc * 2; long Nvf = (long)Gf * Gf * Gf;
    int Cdec = wdc.Co; int Ccat = Cdec + Cskip;
    conv(y, wdc, Gc, Gf, 2, 0, 1);                         // 8 tap-GEMMs
    bn(dcp, Nvf, Cdec, SA, Ccat, 0, 1, nullptr, 0);
    copy_ch_k<<<(int)((Nvf * Cskip + 255) / 256), 256, 0, stream>>>(
        skip, Cskip, SA, Ccat, Cdec, Nvf, Cskip);
    res(SA, Gf, Ccat, Cdec, w11, c11, w12, c12, &w1ds, &c1ds, SD);
    res(SD, Gf, Cdec, Cdec, w21, c21, w22, c22, nullptr, nullptr, outb);
  };

  // ---- forward pass ----
  cvt_in_k<<<((long)V48 * 32 + 255) / 256, 256, 0, stream>>>(x, A0, V48);
  conv(A0, wstem0, 48, 48, 1, 1, 0); bn(stem0, V48, 32, SB, 32, 0, 1, nullptr, 0);
  conv(SB, wstem1, 48, 48, 1, 1, 0); bn(stem1, V48, 32, X0, 32, 0, 1, nullptr, 0);

  enc(X0, 48, 32, 32,  ws1dn, s1dn, ws1r1c1, s1r1c1, ws1r1c2, s1r1c2, nullptr, nullptr,
      ws1r2c1, s1r2c1, ws1r2c2, s1r2c2, X1);
  enc(X1, 24, 32, 64,  ws2dn, s2dn, ws2r1c1, s2r1c1, ws2r1c2, s2r1c2, &ws2r1ds, &s2r1ds,
      ws2r2c1, s2r2c1, ws2r2c2, s2r2c2, X2);
  enc(X2, 12, 64, 128, ws3dn, s3dn, ws3r1c1, s3r1c1, ws3r1c2, s3r1c2, &ws3r1ds, &s3r1ds,
      ws3r2c1, s3r2c1, ws3r2c2, s3r2c2, X3);
  enc(X3, 6, 128, 256, ws4dn, s4dn, ws4r1c1, s4r1c1, ws4r1c2, s4r1c2, &ws4r1ds, &s4r1ds,
      ws4r2c1, s4r2c1, ws4r2c2, s4r2c2, X4);

  dec(X4, 3,  X3, 128, wu1dc, u1dc, wu1r1c1, u1r1c1, wu1r1c2, u1r1c2, wu1r1ds, u1r1ds,
      wu1r2c1, u1r2c1, wu1r2c2, u1r2c2, SE);
  dec(SE, 6,  X2, 64,  wu2dc, u2dc, wu2r1c1, u2r1c1, wu2r1c2, u2r1c2, wu2r1ds, u2r1ds,
      wu2r2c1, u2r2c1, wu2r2c2, u2r2c2, SE);
  dec(SE, 12, X1, 32,  wu3dc, u3dc, wu3r1c1, u3r1c1, wu3r1c2, u3r1c2, wu3r1ds, u3r1ds,
      wu3r2c1, u3r2c1, wu3r2c2, u3r2c2, SE);
  dec(SE, 24, X0, 32,  wu4dc, u4dc, wu4r1c1, u4r1c1, wu4r1c2, u4r1c2, wu4r1ds, u4r1ds,
      wu4r2c1, u4r2c1, wu4r2c2, u4r2c2, SE);

  head_k<<<(V48 + 255) / 256, 256, 0, stream>>>(SE, head_w1, head_b1, head_w2,
                                                head_b2, (float*)d_out, V48, 96);
}
